// FFTSSMBlock_52080773432011
// MI455X (gfx1250) — matrix operand order repaired and checked by reading, not compiled, not run on hardware
//
#include <hip/hip_runtime.h>
#include <hip/hip_bf16.h>
#include <math.h>

// ---------------------------------------------------------------------------
// FFT-SSM block for MI455X (gfx1250, wave32, WMMA).
// GEMMs (in_proj / x_proj / spectral-mix / out_proj) run on v_wmma_f32_16x16x32_bf16.
// out_proj additionally stages its weight tile through the Tensor Data Mover
// (tensor_load_to_lds + s_wait_tensorcnt), the CDNA5 async-tensor path.
// Selective scan: one (b,d) pair per 16-lane half-wave, shfl_xor state reduce.
// ---------------------------------------------------------------------------

typedef __attribute__((ext_vector_type(16))) __bf16 v16bf;
typedef __attribute__((ext_vector_type(8)))  float  v8f;
typedef __attribute__((ext_vector_type(4)))  unsigned int u32x4;
typedef __attribute__((ext_vector_type(8)))  int i32x8;
typedef __attribute__((ext_vector_type(4)))  int i32x4;

#define DEV static __device__ __forceinline__

constexpr int C_     = 128;
constexpr int DIN    = 256;          // d_inner
constexpr int BSZ    = 4;
constexpr int HH     = 64;
constexpr int WW     = 64;
constexpr int LL     = HH * WW;      // 4096
constexpr int MM     = BSZ * LL;     // 16384
constexpr int WRQ    = WW / 2 + 1;   // 33
constexpr int MF     = BSZ * HH * WRQ; // 8448 (divisible by 64)
constexpr int XDP    = 48;           // x_dbl padded 40 -> 48

// workspace offsets (floats)
constexpr size_t OFF_XZ  = 0;                                   // 16384*512
constexpr size_t OFF_XS  = OFF_XZ  + (size_t)MM * 512;          // 16384*256
constexpr size_t OFF_DT  = OFF_XS  + (size_t)MM * DIN;
constexpr size_t OFF_YS  = OFF_DT  + (size_t)MM * DIN;
constexpr size_t OFF_YG  = OFF_YS  + (size_t)MM * DIN;
constexpr size_t OFF_XD  = OFF_YG  + (size_t)MM * DIN;          // 16384*48
constexpr size_t OFF_T1R = OFF_XD  + (size_t)MM * XDP;          // 512*64*33
constexpr size_t OFF_T1I = OFF_T1R + (size_t)BSZ * C_ * HH * WRQ;
constexpr size_t OFF_FQ  = OFF_T1I + (size_t)BSZ * C_ * HH * WRQ; // 8448*256
constexpr size_t OFF_FM  = OFF_FQ  + (size_t)MF * 256;            // 8448*256

constexpr float TWO_PI_OVER_64 = 6.283185307179586f / 64.f;

DEV unsigned short f2bf(float f) {               // fp32 -> bf16 (RNE)
  unsigned u = __builtin_bit_cast(unsigned, f);
  u += 0x7FFFu + ((u >> 16) & 1u);
  return (unsigned short)(u >> 16);
}

union FragBF { v16bf v; unsigned short s[16]; };

// A fragment (16x32 bf16, MxK): lanes 0-15 hold row M=lane, K {0..7,16..23};
// lanes 16-31 hold row M=lane-16, K {8..15,24..31} (ISA 7.12.2).
DEV v16bf load_a_frag(const unsigned short* sA, int K, int i0, int kb, int lane) {
  FragBF f;
  const unsigned short* p = sA + (size_t)(i0 + (lane & 15)) * K + kb + ((lane & 16) ? 8 : 0);
#pragma unroll
  for (int v = 0; v < 8; ++v) {
    int base = (v < 4) ? (2 * v) : (16 + 2 * (v - 4));
    f.s[2 * v]     = p[base];
    f.s[2 * v + 1] = p[base + 1];
  }
  return f.v;
}

// B fragment (32x16 bf16, KxN): lanes 0-15 col N=lane, K {0..7,16..23}; lanes 16-31 K {8..15,24..31}.
DEV v16bf load_b_frag(const unsigned short* sB, int K, int kb, int lane) {
  FragBF f;
  const unsigned short* p = sB + (size_t)(lane & 15) * K + kb + ((lane & 16) ? 8 : 0);
#pragma unroll
  for (int v = 0; v < 8; ++v) { int base = (v < 4) ? (2 * v) : (16 + 2 * (v - 4)); f.s[2 * v] = p[base]; f.s[2 * v + 1] = p[base + 1]; }
  return f.v;
}

// Issue a 2-D TDM load of `rows` x `cols` fp32 tile (row stride = tensorCols
// elements, tensor has tensorRows rows) from `gsrc` into LDS at `lds_off`.
// Descriptor layout per CDNA5 ISA 8.3/8.4. Call from a single wave.
DEV void tdm_load_tile_f32(unsigned lds_off, const float* gsrc,
                           int cols, int rows, int tensorCols, int tensorRows) {
  unsigned long long ga = (unsigned long long)(size_t)gsrc;
  u32x4 g0;
  g0[0] = 1u;                                              // count=1, user descriptor
  g0[1] = lds_off;                                         // lds_addr
  g0[2] = (unsigned)(ga & 0xFFFFFFFFu);                    // global_addr[31:0]
  g0[3] = (unsigned)((ga >> 32) & 0x01FFFFFFu) | (2u << 30); // addr[56:32] | type=2
  i32x8 g1;
  g1[0] = (2 << 16);                  // data_size = 2 (4 bytes); no multicast/pad
  g1[1] = (tensorCols & 0xFFFF) << 16; // tensor_dim0[15:0]  (bits 63:48)
  g1[2] = ((tensorCols >> 16) & 0xFFFF) | ((tensorRows & 0xFFFF) << 16); // dim0 hi | dim1 lo
  g1[3] = ((tensorRows >> 16) & 0xFFFF) | ((cols & 0xFFFF) << 16);       // dim1 hi | tile_dim0
  g1[4] = rows & 0xFFFF;              // tile_dim1 (tile_dim2 = 0)
  g1[5] = tensorCols;                 // tensor_dim0_stride[31:0]
  g1[6] = 0;                          // stride hi | tensor_dim1_stride lo
  g1[7] = 0;
  i32x4 z4 = {0, 0, 0, 0};
#if defined(__clang_major__) && (__clang_major__ >= 23)
  i32x8 z8 = {0, 0, 0, 0, 0, 0, 0, 0};
  __builtin_amdgcn_tensor_load_to_lds(g0, g1, z4, z4, z8, 0);
#else
  __builtin_amdgcn_tensor_load_to_lds(g0, g1, z4, z4, 0);
#endif
}

// ---------------------------------------------------------------------------
// 1) spatial 3x3 conv, writes out with '='
// ---------------------------------------------------------------------------
__global__ __launch_bounds__(64) void k_spatial(const float* __restrict__ x,
                                                const float* __restrict__ wt,
                                                const float* __restrict__ bias,
                                                float* __restrict__ out) {
  int w   = threadIdx.x;
  int bcr = blockIdx.x;
  int r   = bcr & 63;
  int co  = (bcr >> 6) & 127;
  int b   = bcr >> 13;
  float acc = bias[co];
  const float* xb = x  + (size_t)b * 128 * 4096;
  const float* wb = wt + (size_t)co * 128 * 9;
  for (int ci = 0; ci < 128; ++ci) {
    const float* xp = xb + (size_t)ci * 4096;
    const float* wp = wb + ci * 9;
    __builtin_prefetch(xp + 4096);          // next channel plane -> global_prefetch_b8
#pragma unroll
    for (int dr = -1; dr <= 1; ++dr) {
      int rr = r + dr;
      if (rr < 0 || rr > 63) continue;
#pragma unroll
      for (int dc = -1; dc <= 1; ++dc) {
        int cc = w + dc;
        if (cc < 0 || cc > 63) continue;
        acc += xp[rr * 64 + cc] * wp[(dr + 1) * 3 + (dc + 1)];
      }
    }
  }
  out[(size_t)(b * 128 + co) * 4096 + r * 64 + w] = acc;
}

// ---------------------------------------------------------------------------
// 2) in_proj: xz[m][512] = u[m][128] @ W^T ; u is the NCHW->NLC transpose of x
// ---------------------------------------------------------------------------
__global__ __launch_bounds__(128) void k_in_proj(const float* __restrict__ x,
                                                 const float* __restrict__ w,   // [512][128]
                                                 float* __restrict__ xz) {
  __shared__ unsigned short sA[64 * 128];
  __shared__ unsigned short sB[16 * 128];
  const int tid = threadIdx.x;
  const int m0  = blockIdx.x * 64;
  const int n0  = blockIdx.y * 16;
  for (int idx = tid; idx < 64 * 128; idx += 128) {   // coalesced along hw
    int i = idx & 63, k = idx >> 6;
    int m = m0 + i, b = m >> 12, hw = m & 4095;
    sA[i * 128 + k] = f2bf(x[((size_t)(b * 128 + k)) * 4096 + hw]);
  }
  for (int idx = tid; idx < 16 * 128; idx += 128) {
    int n = idx >> 7, k = idx & 127;
    sB[idx] = f2bf(w[(n0 + n) * 128 + k]);
  }
  __syncthreads();
  const int lane = tid & 31;
  const int i0   = (tid >> 5) * 16;
  v8f c = {};
#pragma unroll
  for (int kb = 0; kb < 128; kb += 32) {
    v16bf a = load_a_frag(sA, 128, i0, kb, lane);
    v16bf b = load_b_frag(sB, 128, kb, lane);
    c = __builtin_amdgcn_wmma_f32_16x16x32_bf16(false, a, false, b, (short)0, c, false, false);
  }
  const int col = lane & 15, rb = (lane & 16) ? 8 : 0;
#pragma unroll
  for (int r = 0; r < 8; ++r) {
    int m = m0 + i0 + rb + r;
    xz[(size_t)m * 512 + n0 + col] = c[r];
  }
}

// ---------------------------------------------------------------------------
// 3) depthwise causal conv1d (K=4) + SiLU
// ---------------------------------------------------------------------------
__global__ __launch_bounds__(256) void k_conv1d(const float* __restrict__ xz,
                                                const float* __restrict__ w,   // [256][1][4]
                                                const float* __restrict__ b,
                                                float* __restrict__ xs) {
  int idx = blockIdx.x * 256 + threadIdx.x;
  int d = idx & 255;
  int m = idx >> 8;
  int l = m & 4095;
  int base = m - l;
  float acc = b[d];
#pragma unroll
  for (int j = 0; j < 4; ++j) {
    int lj = l + j - 3;
    if (lj >= 0) acc += w[d * 4 + j] * xz[(size_t)(base + lj) * 512 + d];
  }
  xs[(size_t)m * 256 + d] = acc / (1.f + __expf(-acc));   // silu
}

// ---------------------------------------------------------------------------
// 4) x_proj: xd[m][48] = xs[m][256] @ Wxp^T (N=40 padded to 48 with zero rows)
// ---------------------------------------------------------------------------
__global__ __launch_bounds__(128) void k_x_proj(const float* __restrict__ xs,
                                                const float* __restrict__ w,   // [40][256]
                                                float* __restrict__ xd) {
  __shared__ unsigned short sA[64 * 256];
  __shared__ unsigned short sB[16 * 256];
  const int tid = threadIdx.x;
  const int m0  = blockIdx.x * 64;
  const int n0  = blockIdx.y * 16;
  for (int idx = tid; idx < 64 * 256; idx += 128) {
    int i = idx >> 8, k = idx & 255;
    sA[idx] = f2bf(xs[(size_t)(m0 + i) * 256 + k]);
  }
  for (int idx = tid; idx < 16 * 256; idx += 128) {
    int n = idx >> 8, k = idx & 255;
    int nn = n0 + n;
    sB[idx] = (nn < 40) ? f2bf(w[nn * 256 + k]) : (unsigned short)0;
  }
  __syncthreads();
  const int lane = tid & 31;
  const int i0   = (tid >> 5) * 16;
  v8f c = {};
#pragma unroll
  for (int kb = 0; kb < 256; kb += 32) {
    v16bf a = load_a_frag(sA, 256, i0, kb, lane);
    v16bf b = load_b_frag(sB, 256, kb, lane);
    c = __builtin_amdgcn_wmma_f32_16x16x32_bf16(false, a, false, b, (short)0, c, false, false);
  }
  const int col = lane & 15, rb = (lane & 16) ? 8 : 0;
#pragma unroll
  for (int r = 0; r < 8; ++r) {
    int m = m0 + i0 + rb + r;
    xd[(size_t)m * XDP + n0 + col] = c[r];
  }
}

// ---------------------------------------------------------------------------
// 5) dt = softplus(xd[:, :8] @ Wdt^T + bdt)   (K=8, plain VALU)
// ---------------------------------------------------------------------------
__global__ __launch_bounds__(256) void k_dt(const float* __restrict__ xd,
                                            const float* __restrict__ w,   // [256][8]
                                            const float* __restrict__ b,
                                            float* __restrict__ dt) {
  int idx = blockIdx.x * 256 + threadIdx.x;
  int d = idx & 255;
  int m = idx >> 8;
  const float* xr = xd + (size_t)m * XDP;
  float s = b[d];
#pragma unroll
  for (int r = 0; r < 8; ++r) s += xr[r] * w[d * 8 + r];
  dt[(size_t)m * 256 + d] = (s > 20.f) ? s : log1pf(__expf(s));
}

// ---------------------------------------------------------------------------
// 6) selective scan: 16 lanes (states) per (b,d); serial over L=4096
// ---------------------------------------------------------------------------
__global__ __launch_bounds__(256) void k_scan(const float* __restrict__ xs,
                                              const float* __restrict__ dt,
                                              const float* __restrict__ xd,
                                              const float* __restrict__ alog,
                                              const float* __restrict__ dvec,
                                              float* __restrict__ ys) {
  int tid = threadIdx.x;
  int grp = blockIdx.x * 16 + (tid >> 4);   // 0..1023 = (b,d)
  int n   = tid & 15;                       // state index
  int b   = grp >> 8;
  int d   = grp & 255;
  float A  = -__expf(alog[d * 16 + n]);
  float Dv = dvec[d];
  float h  = 0.f;
  for (int t = 0; t < LL; ++t) {
    int m = b * LL + t;
    float xt  = xs[(size_t)m * 256 + d];
    float dtt = dt[(size_t)m * 256 + d];
    const float* xr = xd + (size_t)m * XDP;
    float Bt = xr[8 + n];
    float Ct = xr[24 + n];
    float dA = __expf(dtt * A);
    h = dA * h + dtt * Bt * xt;
    float y = h * Ct;
    y += __shfl_xor(y, 1, 16);
    y += __shfl_xor(y, 2, 16);
    y += __shfl_xor(y, 4, 16);
    y += __shfl_xor(y, 8, 16);
    if (n == 0) ys[(size_t)m * 256 + d] = y + xt * Dv;
  }
}

// ---------------------------------------------------------------------------
// 7) gate: yg = ys * silu(z)
// ---------------------------------------------------------------------------
__global__ __launch_bounds__(256) void k_gate(const float* __restrict__ ys,
                                              const float* __restrict__ xz,
                                              float* __restrict__ yg) {
  int idx = blockIdx.x * 256 + threadIdx.x;
  int d = idx & 255;
  int m = idx >> 8;
  float z = xz[(size_t)m * 512 + 256 + d];
  yg[(size_t)m * 256 + d] = ys[(size_t)m * 256 + d] * (z / (1.f + __expf(-z)));
}

// ---------------------------------------------------------------------------
// 8) out_proj: out[b,o,hw] += yg[m][256] @ Wout^T
//    Weight tile (16x256 fp32) is DMA'd into LDS by the Tensor Data Mover.
// ---------------------------------------------------------------------------
__global__ __launch_bounds__(128) void k_out_proj(const float* __restrict__ yg,
                                                  const float* __restrict__ w,  // [128][256]
                                                  float* __restrict__ out) {
  __shared__ unsigned short sA[64 * 256];
  __shared__ unsigned short sB[16 * 256];
  __shared__ float sB32[16 * 256];
  const int tid = threadIdx.x;
  const int m0  = blockIdx.x * 64;
  const int n0  = blockIdx.y * 16;

  if (tid < 32) {        // wave 0 issues the TDM tile load and waits for it
    unsigned lds_off = (unsigned)(size_t)(void*)sB32;
    tdm_load_tile_f32(lds_off, w + (size_t)n0 * 256,
                      /*cols=*/256, /*rows=*/16, /*tensorCols=*/256, /*tensorRows=*/128);
    __builtin_amdgcn_s_wait_tensorcnt(0);
  }
  for (int idx = tid; idx < 64 * 256; idx += 128) {
    int i = idx >> 8, k = idx & 255;
    sA[idx] = f2bf(yg[(size_t)(m0 + i) * 256 + k]);
  }
  __syncthreads();                       // TDM data + sA visible to all waves
  for (int idx = tid; idx < 16 * 256; idx += 128)
    sB[idx] = f2bf(sB32[idx]);           // fp32 tile -> bf16 fragments
  __syncthreads();

  const int lane = tid & 31;
  const int i0   = (tid >> 5) * 16;
  v8f c = {};
#pragma unroll
  for (int kb = 0; kb < 256; kb += 32) {
    v16bf a = load_a_frag(sA, 256, i0, kb, lane);
    v16bf b = load_b_frag(sB, 256, kb, lane);
    c = __builtin_amdgcn_wmma_f32_16x16x32_bf16(false, a, false, b, (short)0, c, false, false);
  }
  const int col = lane & 15, rb = (lane & 16) ? 8 : 0;
#pragma unroll
  for (int r = 0; r < 8; ++r) {
    int m  = m0 + i0 + rb + r;
    int bb = m >> 12, hw = m & 4095;
    float* p = out + (size_t)(bb * 128 + n0 + col) * 4096 + hw;
    *p += c[r];
  }
}

// ---------------------------------------------------------------------------
// 9-13) spectral branch: rfft rows -> fft cols -> 1x1 mix (WMMA) -> ifft cols -> irfft rows
// ---------------------------------------------------------------------------
__global__ __launch_bounds__(256) void k_rowfft(const float* __restrict__ x,
                                                float* __restrict__ t1r,
                                                float* __restrict__ t1i) {
  int idx = blockIdx.x * 256 + threadIdx.x;   // < 32768*33
  int wr  = idx % 33;
  int row = idx / 33;                          // bc*64 + h
  const float* xp = x + (size_t)row * 64;
  float sr = 0.f, si = 0.f;
  float step = TWO_PI_OVER_64 * (float)wr;
  for (int w = 0; w < 64; ++w) {
    float sn, cs; __sincosf(step * (float)w, &sn, &cs);
    float xv = xp[w];
    sr += xv * cs;
    si -= xv * sn;
  }
  t1r[idx] = sr;
  t1i[idx] = si;
}

__global__ __launch_bounds__(256) void k_colfft(const float* __restrict__ t1r,
                                                const float* __restrict__ t1i,
                                                float* __restrict__ fq) {
  int idx  = blockIdx.x * 256 + threadIdx.x;  // < 512*33*64
  int hk   = idx & 63;
  int rest = idx >> 6;
  int wr   = rest % 33;
  int bc   = rest / 33;
  int b = bc >> 7, c = bc & 127;
  float sr = 0.f, si = 0.f;
  float step = TWO_PI_OVER_64 * (float)hk;
  for (int h = 0; h < 64; ++h) {
    size_t o = (size_t)(bc * 64 + h) * 33 + wr;
    float fr = t1r[o], fi = t1i[o];
    float sn, cs; __sincosf(step * (float)h, &sn, &cs);
    sr += fr * cs + fi * sn;      // *(e^{-i theta})
    si += fi * cs - fr * sn;
  }
  size_t m = (size_t)(b * 64 + hk) * 33 + wr;  // GEMM row
  fq[m * 256 + c]       = sr;
  fq[m * 256 + 128 + c] = si;
}

__global__ __launch_bounds__(128) void k_mix(const float* __restrict__ fq,
                                             const float* __restrict__ w,   // [256][256]
                                             const float* __restrict__ bias,
                                             float* __restrict__ fm) {
  __shared__ unsigned short sA[64 * 256];
  __shared__ unsigned short sB[16 * 256];
  const int tid = threadIdx.x;
  const int m0  = blockIdx.x * 64;
  const int n0  = blockIdx.y * 16;
  for (int idx = tid; idx < 64 * 256; idx += 128) {
    int i = idx >> 8, k = idx & 255;
    sA[idx] = f2bf(fq[(size_t)(m0 + i) * 256 + k]);
  }
  for (int idx = tid; idx < 16 * 256; idx += 128) {
    int n = idx >> 8, k = idx & 255;
    sB[idx] = f2bf(w[(n0 + n) * 256 + k]);
  }
  __syncthreads();
  const int lane = tid & 31;
  const int i0   = (tid >> 5) * 16;
  v8f c = {};
#pragma unroll
  for (int kb = 0; kb < 256; kb += 32) {
    v16bf a = load_a_frag(sA, 256, i0, kb, lane);
    v16bf b = load_b_frag(sB, 256, kb, lane);
    c = __builtin_amdgcn_wmma_f32_16x16x32_bf16(false, a, false, b, (short)0, c, false, false);
  }
  const int col = lane & 15, rb = (lane & 16) ? 8 : 0;
#pragma unroll
  for (int r = 0; r < 8; ++r) {
    int m = m0 + i0 + rb + r;
    fm[(size_t)m * 256 + n0 + col] = c[r] + bias[n0 + col];
  }
}

__global__ __launch_bounds__(256) void k_icol(const float* __restrict__ fm,
                                              float* __restrict__ t1r,
                                              float* __restrict__ t1i) {
  int idx  = blockIdx.x * 256 + threadIdx.x;  // < 512*64*33
  int wr   = idx % 33;
  int rest = idx / 33;
  int h    = rest & 63;
  int bc   = rest >> 6;
  int b = bc >> 7, c = bc & 127;
  float sr = 0.f, si = 0.f;
  float step = TWO_PI_OVER_64 * (float)h;
  for (int hk = 0; hk < 64; ++hk) {
    const float* p = fm + ((size_t)((b * 64 + hk) * 33 + wr)) * 256;
    float fr = p[c], fi = p[128 + c];
    float sn, cs; __sincosf(step * (float)hk, &sn, &cs);
    sr += fr * cs - fi * sn;      // *(e^{+i theta})
    si += fr * sn + fi * cs;
  }
  size_t o = (size_t)(bc * 64 + h) * 33 + wr;
  t1r[o] = sr;
  t1i[o] = si;
}

__global__ __launch_bounds__(256) void k_irow(const float* __restrict__ t1r,
                                              const float* __restrict__ t1i,
                                              float* __restrict__ out) {
  int idx = blockIdx.x * 256 + threadIdx.x;   // < 512*4096
  int w   = idx & 63;
  int h   = (idx >> 6) & 63;
  int bc  = idx >> 12;
  const float* pr = t1r + ((size_t)bc * 64 + h) * 33;
  const float* pi = t1i + ((size_t)bc * 64 + h) * 33;
  float acc  = pr[0];
  float step = TWO_PI_OVER_64 * (float)w;
  for (int wr = 1; wr < 32; ++wr) {
    float sn, cs; __sincosf(step * (float)wr, &sn, &cs);
    acc += 2.f * (pr[wr] * cs - pi[wr] * sn);
  }
  {
    float sn, cs; __sincosf(step * 32.f, &sn, &cs);
    acc += pr[32] * cs - pi[32] * sn;        // Nyquist bin, weight 1
  }
  out[idx] += acc * (1.f / 4096.f);          // irfft2 backward norm 1/(H*W)
}

// ---------------------------------------------------------------------------
extern "C" void kernel_launch(void* const* d_in, const int* in_sizes, int n_in,
                              void* d_out, int out_size, void* d_ws, size_t ws_size,
                              hipStream_t stream) {
  (void)in_sizes; (void)n_in; (void)out_size; (void)ws_size;
  const float* x      = (const float*)d_in[0];
  const float* w_sp   = (const float*)d_in[1];
  const float* b_sp   = (const float*)d_in[2];
  const float* w_spec = (const float*)d_in[3];
  const float* b_spec = (const float*)d_in[4];
  const float* w_in   = (const float*)d_in[5];
  const float* w_c1   = (const float*)d_in[6];
  const float* b_c1   = (const float*)d_in[7];
  const float* w_xp   = (const float*)d_in[8];
  const float* w_dt   = (const float*)d_in[9];
  const float* b_dt   = (const float*)d_in[10];
  const float* alog   = (const float*)d_in[11];
  const float* dvec   = (const float*)d_in[12];
  const float* w_out  = (const float*)d_in[13];
  float* out = (float*)d_out;
  float* ws  = (float*)d_ws;

  float* xz  = ws + OFF_XZ;
  float* xs  = ws + OFF_XS;
  float* dt  = ws + OFF_DT;
  float* ys  = ws + OFF_YS;
  float* yg  = ws + OFF_YG;
  float* xd  = ws + OFF_XD;
  float* t1r = ws + OFF_T1R;
  float* t1i = ws + OFF_T1I;
  float* fq  = ws + OFF_FQ;
  float* fm  = ws + OFF_FM;

  // spatial conv writes '=' -> must be first
  k_spatial<<<dim3(BSZ * C_ * HH), dim3(WW), 0, stream>>>(x, w_sp, b_sp, out);

  // spectral branch (+= into out)
  k_rowfft<<<dim3(4224), dim3(256), 0, stream>>>(x, t1r, t1i);
  k_colfft<<<dim3(4224), dim3(256), 0, stream>>>(t1r, t1i, fq);
  k_mix<<<dim3(MF / 64, 16), dim3(128), 0, stream>>>(fq, w_spec, b_spec, fm);
  k_icol<<<dim3(4224), dim3(256), 0, stream>>>(fm, t1r, t1i);
  k_irow<<<dim3(8192), dim3(256), 0, stream>>>(t1r, t1i, out);

  // mamba branch (+= into out at the end)
  k_in_proj<<<dim3(MM / 64, 32), dim3(128), 0, stream>>>(x, w_in, xz);
  k_conv1d<<<dim3(16384), dim3(256), 0, stream>>>(xz, w_c1, b_c1, xs);
  k_x_proj<<<dim3(MM / 64, 3), dim3(128), 0, stream>>>(xs, w_xp, xd);
  k_dt<<<dim3(16384), dim3(256), 0, stream>>>(xd, w_dt, b_dt, dt);
  k_scan<<<dim3(64), dim3(256), 0, stream>>>(xs, dt, xd, alog, dvec, ys);
  k_gate<<<dim3(16384), dim3(256), 0, stream>>>(ys, xz, yg);
  k_out_proj<<<dim3(MM / 64, 8), dim3(128), 0, stream>>>(yg, w_out, out);
}
